// CausalPerformer_5918464934283
// MI455X (gfx1250) — matrix operand order repaired and checked by reading, not compiled, not run on hardware
//
#include <hip/hip_runtime.h>

// ---------------------------------------------------------------------------
// Types for WMMA
// ---------------------------------------------------------------------------
typedef __attribute__((ext_vector_type(16))) __bf16 v16bf;
typedef __attribute__((ext_vector_type(8)))  float  v8f;

#define EPS 1e-6f

// fp32 -> bf16 (round to nearest even)
__device__ __forceinline__ unsigned short f2bf(float x) {
    unsigned u = __float_as_uint(x);
    unsigned r = (u + 0x7FFFu + ((u >> 16) & 1u)) >> 16;
    return (unsigned short)r;
}

// ---------------------------------------------------------------------------
// GEMM: C[M,N] = A[M,K] * B[N,K]^T   (A,B,C all row-major fp32; bf16 WMMA)
// Block tile 128x128, K-step 32, 256 threads = 8 waves, wave tile 64x32.
// Register double-buffering: global loads for tile k+1 issued before the
// WMMA compute of tile k; global_prefetch primes tile k+2.
// ---------------------------------------------------------------------------
#define TILE_M 128
#define TILE_N 128
#define TILE_K 32
#define LDSTRIDE 40   // bf16 elems per LDS row: 32 + 8 pad = 80 bytes (16B aligned)

__global__ __launch_bounds__(256)
void gemm_abt_bf16wmma(const float* __restrict__ A,
                       const float* __restrict__ Bw,
                       float* __restrict__ C,
                       int M, int N, int K) {
    __shared__ unsigned short lA[TILE_M * LDSTRIDE];
    __shared__ unsigned short lB[TILE_N * LDSTRIDE];

    const int tid  = threadIdx.x;
    const int lane = tid & 31;
    const int wave = tid >> 5;
    const int waveM = (wave & 1) * 64;   // 2 waves along M
    const int waveN = (wave >> 1) * 32;  // 4 waves along N

    const int blockM = blockIdx.x * TILE_M;
    const int blockN = blockIdx.y * TILE_N;

    // staging: each thread converts 16 consecutive f32 of one row
    const int ldRow = tid >> 1;         // 0..127
    const int ldCol = (tid & 1) * 16;   // 0 or 16

    const float* aRow = A  + (size_t)(blockM + ldRow) * K + ldCol;
    const float* bRow = Bw + (size_t)(blockN + ldRow) * K + ldCol;

    v8f acc[4][2];
    const v8f vzero = {0.f,0.f,0.f,0.f,0.f,0.f,0.f,0.f};
    #pragma unroll
    for (int i = 0; i < 4; ++i)
        #pragma unroll
        for (int j = 0; j < 2; ++j)
            acc[i][j] = vzero;

    const int mhalf = lane >> 4;    // 0/1 (which K-half this lane holds)
    const int mrow  = lane & 15;    // row/col within 16

    union Frag { uint4 q[2]; v16bf v; };

    // ---- prologue: fetch tile 0 into registers ----
    float4 ra[4], rb[4];
    {
        const float4* sa = (const float4*)(aRow);
        const float4* sb = (const float4*)(bRow);
        #pragma unroll
        for (int i = 0; i < 4; ++i) { ra[i] = sa[i]; rb[i] = sb[i]; }
    }

    for (int k0 = 0; k0 < K; k0 += TILE_K) {
        __syncthreads();   // all waves done reading previous tile
        // ---- convert current tile regs -> LDS (bf16) ----
        {
            unsigned int* dA = (unsigned int*)&lA[ldRow * LDSTRIDE + ldCol];
            unsigned int* dB = (unsigned int*)&lB[ldRow * LDSTRIDE + ldCol];
            #pragma unroll
            for (int i = 0; i < 4; ++i) {
                float4 x = ra[i];
                dA[2*i+0] = (unsigned)f2bf(x.x) | ((unsigned)f2bf(x.y) << 16);
                dA[2*i+1] = (unsigned)f2bf(x.z) | ((unsigned)f2bf(x.w) << 16);
                float4 y = rb[i];
                dB[2*i+0] = (unsigned)f2bf(y.x) | ((unsigned)f2bf(y.y) << 16);
                dB[2*i+1] = (unsigned)f2bf(y.z) | ((unsigned)f2bf(y.w) << 16);
            }
        }
        __syncthreads();

        // ---- issue global loads for tile k+1 (in flight during compute) ----
        const int kn = k0 + TILE_K;
        if (kn < K) {
            const float4* sa = (const float4*)(aRow + kn);
            const float4* sb = (const float4*)(bRow + kn);
            #pragma unroll
            for (int i = 0; i < 4; ++i) { ra[i] = sa[i]; rb[i] = sb[i]; }
            // prime L2 for tile k+2 (emits global_prefetch_b8)
            if (kn + TILE_K < K) {
                __builtin_prefetch(aRow + kn + TILE_K, 0, 1);
                __builtin_prefetch(bRow + kn + TILE_K, 0, 1);
            }
        }

        // ---- B fragments: 32x16 (KxN); lane = N (0..15), K-half by lane>=16 ----
        Frag bf[2];
        #pragma unroll
        for (int tn = 0; tn < 2; ++tn) {
            const unsigned short* p = &lB[(waveN + tn*16 + mrow) * LDSTRIDE + mhalf*8];
            bf[tn].q[0] = *(const uint4*)(p + 0);
            bf[tn].q[1] = *(const uint4*)(p + 16);
        }

        // ---- A fragments + WMMA: 16x32 per tile ----
        #pragma unroll
        for (int tm = 0; tm < 4; ++tm) {
            Frag af;
            const unsigned short* p = &lA[(waveM + tm*16 + mrow) * LDSTRIDE + mhalf*8];
            af.q[0] = *(const uint4*)(p + 0);    // K = mhalf*8 .. +7
            af.q[1] = *(const uint4*)(p + 16);   // K = 16 + mhalf*8 .. +7
            #pragma unroll
            for (int tn = 0; tn < 2; ++tn) {
                acc[tm][tn] = __builtin_amdgcn_wmma_f32_16x16x32_bf16(
                    false, af.v, false, bf[tn].v, (short)0, acc[tm][tn], false, false);
            }
        }
    }

    // ---- store C (16x16 f32 C/D layout: lane 0-15 -> M=j, lane 16-31 -> M=8+j, N=lane&15)
    #pragma unroll
    for (int tm = 0; tm < 4; ++tm) {
        #pragma unroll
        for (int tn = 0; tn < 2; ++tn) {
            const int mBase = blockM + waveM + tm*16 + mhalf*8;
            const int nCol  = blockN + waveN + tn*16 + mrow;
            #pragma unroll
            for (int j = 0; j < 8; ++j) {
                C[(size_t)(mBase + j) * N + nCol] = acc[tm][tn][j];
            }
        }
    }
}

// ---------------------------------------------------------------------------
// Random-feature maps: one thread per (row, head). 7 dots of length 64,
// exp(-x^2/2), normalize; output stride padded to 8.
// ---------------------------------------------------------------------------
__global__ __launch_bounds__(256)
void featmap(const float* __restrict__ Qh, const float* __restrict__ Kh,
             const float* __restrict__ omega,
             float* __restrict__ Qp, float* __restrict__ Kp, int rows) {
    int idx = blockIdx.x * blockDim.x + threadIdx.x;    // row*16 + h
    if (idx >= rows * 16) return;
    int r = idx >> 4, h = idx & 15;
    const float* q = Qh + (size_t)r * 1024 + h * 64;
    const float* k = Kh + (size_t)r * 1024 + h * 64;
    float qp[7], kp[7], qs = 0.f, ks = 0.f;
    #pragma unroll
    for (int f = 0; f < 7; ++f) {
        const float* om = omega + f * 64;
        float xq = 0.f, xk = 0.f;
        #pragma unroll 8
        for (int d = 0; d < 64; ++d) { xq += q[d] * om[d]; xk += k[d] * om[d]; }
        qp[f] = __expf(-0.5f * xq * xq);
        kp[f] = __expf(-0.5f * xk * xk);
        qs += qp[f]; ks += kp[f];
    }
    float qi = 1.f / (qs + EPS), ki = 1.f / (ks + EPS);
    float* qo = Qp + (size_t)idx * 8;
    float* ko = Kp + (size_t)idx * 8;
    #pragma unroll
    for (int f = 0; f < 7; ++f) { qo[f] = qp[f] * qi; ko[f] = kp[f] * ki; }
    qo[7] = 0.f; ko[7] = 0.f;
}

// ---------------------------------------------------------------------------
// 3-phase chunked causal scan. NC chunks of CS positions (NC*CS == S).
// One wave per (b*h, chunk); lane owns channels d=lane and d=lane+32.
// ---------------------------------------------------------------------------
#define NC 32
#define CS 128

__global__ __launch_bounds__(32)
void scan_chunk_sum(const float* __restrict__ Kp, const float* __restrict__ Vh,
                    float* __restrict__ CKV, float* __restrict__ CK, int S) {
    const int bh = blockIdx.x;         // b*16 + h
    const int c  = blockIdx.y;
    const int b  = bh >> 4, h = bh & 15;
    const int lane = threadIdx.x;
    float kv0[7], kv1[7], ks[7];
    #pragma unroll
    for (int f = 0; f < 7; ++f) { kv0[f] = 0.f; kv1[f] = 0.f; ks[f] = 0.f; }

    for (int s = c * CS; s < (c + 1) * CS; ++s) {
        const int r = b * S + s;
        const float* kp = Kp + ((size_t)r * 16 + h) * 8;
        const float* v  = Vh + (size_t)r * 1024 + h * 64;
        float v0 = v[lane], v1 = v[lane + 32];
        #pragma unroll
        for (int f = 0; f < 7; ++f) {
            float kf = kp[f];
            kv0[f] += kf * v0; kv1[f] += kf * v1; ks[f] += kf;
        }
    }
    size_t base = ((size_t)bh * NC + c) * 7 * 64;
    #pragma unroll
    for (int f = 0; f < 7; ++f) {
        CKV[base + f*64 + lane]      = kv0[f];
        CKV[base + f*64 + lane + 32] = kv1[f];
    }
    if (lane == 0) {
        size_t kb = ((size_t)bh * NC + c) * 8;
        #pragma unroll
        for (int f = 0; f < 7; ++f) CK[kb + f] = ks[f];
    }
}

__global__ __launch_bounds__(64)
void scan_prefix(float* __restrict__ CKV, float* __restrict__ CK) {
    const int bh = blockIdx.x;
    const int t  = threadIdx.x;     // 64 threads
    #pragma unroll
    for (int i = 0; i < 7; ++i) {
        int fd = t + i * 64;        // 0..447  (= f*64 + d)
        float run = 0.f;
        for (int c = 0; c < NC; ++c) {
            size_t off = ((size_t)bh * NC + c) * 7 * 64 + fd;
            float tmp = CKV[off];
            CKV[off] = run;         // exclusive prefix
            run += tmp;
        }
    }
    if (t < 7) {
        float run = 0.f;
        for (int c = 0; c < NC; ++c) {
            size_t off = ((size_t)bh * NC + c) * 8 + t;
            float tmp = CK[off];
            CK[off] = run;
            run += tmp;
        }
    }
}

__global__ __launch_bounds__(32)
void scan_out(const float* __restrict__ Qp, const float* __restrict__ Kp,
              const float* __restrict__ Vh,
              const float* __restrict__ CKV, const float* __restrict__ CK,
              float* __restrict__ Out, int S) {
    const int bh = blockIdx.x, c = blockIdx.y;
    const int b = bh >> 4, h = bh & 15;
    const int lane = threadIdx.x;

    float kv0[7], kv1[7], ks[7];
    size_t base = ((size_t)bh * NC + c) * 7 * 64;
    size_t kb   = ((size_t)bh * NC + c) * 8;
    #pragma unroll
    for (int f = 0; f < 7; ++f) {
        kv0[f] = CKV[base + f*64 + lane];
        kv1[f] = CKV[base + f*64 + lane + 32];
        ks[f]  = CK[kb + f];
    }

    for (int s = c * CS; s < (c + 1) * CS; ++s) {
        const int r = b * S + s;
        const float* kp = Kp + ((size_t)r * 16 + h) * 8;
        const float* qp = Qp + ((size_t)r * 16 + h) * 8;
        const float* v  = Vh + (size_t)r * 1024 + h * 64;
        float v0 = v[lane], v1 = v[lane + 32];
        float n0 = 0.f, n1 = 0.f, den = 0.f;
        #pragma unroll
        for (int f = 0; f < 7; ++f) {
            float kf = kp[f], qf = qp[f];
            kv0[f] += kf * v0;  kv1[f] += kf * v1;  ks[f] += kf;   // inclusive
            n0 += qf * kv0[f];  n1 += qf * kv1[f];  den += qf * ks[f];
        }
        float inv = 1.f / (den + EPS);
        float* o = Out + (size_t)r * 1024 + h * 64;
        o[lane]      = n0 * inv;
        o[lane + 32] = n1 * inv;
    }
}

// ---------------------------------------------------------------------------
// Launch
// ---------------------------------------------------------------------------
extern "C" void kernel_launch(void* const* d_in, const int* in_sizes, int n_in,
                              void* d_out, int out_size, void* d_ws, size_t ws_size,
                              hipStream_t stream) {
    (void)in_sizes; (void)n_in; (void)out_size; (void)ws_size;

    const float* q     = (const float*)d_in[0];
    const float* k     = (const float*)d_in[1];
    const float* v     = (const float*)d_in[2];
    const float* w_q   = (const float*)d_in[3];
    const float* w_k   = (const float*)d_in[4];
    const float* w_v   = (const float*)d_in[5];
    const float* w_o   = (const float*)d_in[6];
    const float* omega = (const float*)d_in[7];

    const int B = 2, S = 4096, D = 1024;
    const int M = B * S;                    // 8192 rows

    float* ws  = (float*)d_ws;
    float* Qh  = ws;                        // M*D
    float* Kh  = Qh + (size_t)M * D;
    float* Vh  = Kh + (size_t)M * D;
    float* AO  = Vh + (size_t)M * D;        // attention output, M*D
    float* Qp  = AO + (size_t)M * D;        // M*16*8
    float* Kp  = Qp + (size_t)M * 16 * 8;
    float* CKV = Kp + (size_t)M * 16 * 8;   // 32*NC*7*64
    float* CK  = CKV + (size_t)32 * NC * 7 * 64;  // 32*NC*8

    dim3 gg(M / TILE_M, D / TILE_N);        // 64 x 8
    dim3 gb(256);

    gemm_abt_bf16wmma<<<gg, gb, 0, stream>>>(q, w_q, Qh, M, D, D);
    gemm_abt_bf16wmma<<<gg, gb, 0, stream>>>(k, w_k, Kh, M, D, D);
    gemm_abt_bf16wmma<<<gg, gb, 0, stream>>>(v, w_v, Vh, M, D, D);

    featmap<<<(M * 16 + 255) / 256, 256, 0, stream>>>(Qh, Kh, omega, Qp, Kp, M);

    scan_chunk_sum<<<dim3(32, NC), 32, 0, stream>>>(Kp, Vh, CKV, CK, S);
    scan_prefix<<<32, 64, 0, stream>>>(CKV, CK);
    scan_out<<<dim3(32, NC), 32, 0, stream>>>(Qp, Kp, Vh, CKV, CK, AO, S);

    gemm_abt_bf16wmma<<<gg, gb, 0, stream>>>(AO, w_o, (float*)d_out, M, D, D);
}
